// AutoEncoderTopK_12249246728717
// MI455X (gfx1250) — matrix operand order repaired and checked by reading, not compiled, not run on hardware
//
#include <hip/hip_runtime.h>

typedef __attribute__((ext_vector_type(16))) __bf16 v16bf;
typedef __attribute__((ext_vector_type(8)))  __bf16 v8bf;
typedef __attribute__((ext_vector_type(8)))  float  v8f;
typedef __attribute__((ext_vector_type(4)))  int    v4i;
typedef __attribute__((ext_vector_type(4)))  float  f4;
typedef __attribute__((address_space(1)))    v4i    gv4i;   // global int4
typedef __attribute__((address_space(3)))    v4i    lv4i;   // LDS int4

#define ACT_DIM   768
#define DICT_SIZE 24576
#define NROWS     8192
#define TOPK      32
#define BK        32
#define LDA       40   // padded bf16 elems per LDS row (80B, 16B-aligned chunks)

#if __has_builtin(__builtin_amdgcn_global_load_async_to_lds_b128)
#define HAVE_ASYNC_LDS 1
#endif

#ifdef HAVE_ASYNC_LDS
#if __has_builtin(__builtin_amdgcn_s_wait_asynccnt)
#define WAIT_ASYNC(n) __builtin_amdgcn_s_wait_asynccnt(n)
#else
#define WAIT_ASYNC(n) asm volatile("s_wait_asynccnt %0" ::"i"(n) : "memory")
#endif
#else
#define WAIT_ASYNC(n) ((void)0)
#endif

// ---------------- bf16 helpers ----------------
__device__ __forceinline__ unsigned short f32_to_bf16_bits(float f) {
    unsigned u = __builtin_bit_cast(unsigned, f);
    return (unsigned short)((u + 0x7FFFu + ((u >> 16) & 1u)) >> 16);
}
__device__ __forceinline__ float bf16_to_f32(__bf16 h) {
    unsigned short s = __builtin_bit_cast(unsigned short, h);
    return __builtin_bit_cast(float, ((unsigned)s) << 16);
}
__device__ __forceinline__ v16bf cat16(v8bf a, v8bf b) {
    return __builtin_shufflevector(a, b, 0,1,2,3,4,5,6,7,8,9,10,11,12,13,14,15);
}
__device__ __forceinline__ v8f wmma_bf16(v16bf a, v16bf b, v8f c) {
    return __builtin_amdgcn_wmma_f32_16x16x32_bf16(false, a, false, b, (short)0, c, false, false);
}
__device__ __forceinline__ void async_cp16(const void* g, void* l) {
#ifdef HAVE_ASYNC_LDS
    __builtin_amdgcn_global_load_async_to_lds_b128(
        (gv4i*)(void*)g, (lv4i*)l, 0, 0);
#else
    *(uint4*)l = *(const uint4*)g;   // synchronous fallback
#endif
}

// ---------------- Kernel 0: fp32 -> split-bf16 planes (hi/lo) ----------------
__device__ __forceinline__ void split4(const float a[4], uint2* ph, uint2* pl) {
    unsigned short h[4], l[4];
#pragma unroll
    for (int c = 0; c < 4; ++c) {
        h[c] = f32_to_bf16_bits(a[c]);
        float hf = __builtin_bit_cast(float, ((unsigned)h[c]) << 16);
        l[c] = f32_to_bf16_bits(a[c] - hf);
    }
    ph->x = (unsigned)h[0] | ((unsigned)h[1] << 16);
    ph->y = (unsigned)h[2] | ((unsigned)h[3] << 16);
    pl->x = (unsigned)l[0] | ((unsigned)l[1] << 16);
    pl->y = (unsigned)l[2] | ((unsigned)l[3] << 16);
}

__global__ __launch_bounds__(256)
void cvt_split_x(const float* __restrict__ x, const float* __restrict__ b_dec,
                 __bf16* __restrict__ Xh, __bf16* __restrict__ Xl) {
    long i4 = (long)blockIdx.x * 256 + threadIdx.x;     // float4 index
    float4 v = ((const float4*)x)[i4];
    int k = (int)((i4 * 4) % ACT_DIM);
    float4 bd = *(const float4*)&b_dec[k];
    float a[4] = {v.x - bd.x, v.y - bd.y, v.z - bd.z, v.w - bd.w};
    uint2 ph, pl;
    split4(a, &ph, &pl);
    ((uint2*)Xh)[i4] = ph;
    ((uint2*)Xl)[i4] = pl;
}

__global__ __launch_bounds__(256)
void cvt_split_w(const float* __restrict__ w,
                 __bf16* __restrict__ Wh, __bf16* __restrict__ Wl) {
    long i4 = (long)blockIdx.x * 256 + threadIdx.x;
    float4 v = ((const float4*)w)[i4];
    float a[4] = {v.x, v.y, v.z, v.w};
    uint2 ph, pl;
    split4(a, &ph, &pl);
    ((uint2*)Wh)[i4] = ph;
    ((uint2*)Wl)[i4] = pl;
}

// ---------------- Kernel 1a: encoder GEMM, async-LDS staging from bf16 planes ----------------
// 512 threads = 16 waves; block tile 256x128; double-buffered LDS.
#define BM2 256
#define BN2 128
#define A_H 0
#define A_L (BM2 * LDA)                        // 10240
#define B_H (2 * BM2 * LDA)                    // 20480
#define B_L (2 * BM2 * LDA + BN2 * LDA)        // 25600
#define LDSBUF (2 * BM2 * LDA + 2 * BN2 * LDA) // 30720 elems (60KB)

__device__ __forceinline__ void stage_async(
    const __bf16* __restrict__ Xh, const __bf16* __restrict__ Xl,
    const __bf16* __restrict__ Wh, const __bf16* __restrict__ Wl,
    __bf16* buf, int tid, int aRow0, int bRow0, int kt) {
#pragma unroll
    for (int i = 0; i < 2; ++i) {            // A planes: 1024 16B-chunks each
        int id = tid + i * 512;
        int r = id >> 2, c = id & 3;
        long go = (long)(aRow0 + r) * ACT_DIM + kt + c * 8;
        int  lo = r * LDA + c * 8;
        async_cp16(Xh + go, buf + A_H + lo);
        async_cp16(Xl + go, buf + A_L + lo);
    }
    {                                        // B planes: 512 chunks each
        int r = tid >> 2, c = tid & 3;
        long go = (long)(bRow0 + r) * ACT_DIM + kt + c * 8;
        int  lo = r * LDA + c * 8;
        async_cp16(Wh + go, buf + B_H + lo);
        async_cp16(Wl + go, buf + B_L + lo);
    }
}

__global__ __launch_bounds__(512)
void sae_encode_gemm_async(const __bf16* __restrict__ Xh, const __bf16* __restrict__ Xl,
                           const __bf16* __restrict__ Wh, const __bf16* __restrict__ Wl,
                           const float* __restrict__ b_enc, float* __restrict__ pre) {
    __shared__ __bf16 lds[2 * LDSBUF];       // 120 KB

    const int tid  = threadIdx.x;
    const int bn   = blockIdx.x;
    const int bm   = blockIdx.y;
    const int wave = tid >> 5;
    const int lane = tid & 31;
    const int wm   = wave & 7;     // m sub-tile base wm*32  (2 wmma tiles)
    const int wn   = wave >> 3;    // n sub-tile base wn*64  (4 wmma tiles)
    const int lmn  = lane & 15;
    const int kh   = lane >> 4;
    const int aRow0 = bm * BM2, bRow0 = bn * BN2;

    v8f acc[2][4];
#pragma unroll
    for (int i = 0; i < 2; ++i)
#pragma unroll
        for (int j = 0; j < 4; ++j) acc[i][j] = (v8f){0,0,0,0,0,0,0,0};

    stage_async(Xh, Xl, Wh, Wl, lds, tid, aRow0, bRow0, 0);

    for (int kt2 = 0; kt2 < ACT_DIM / BK; ++kt2) {
        __bf16* cur = lds + (kt2 & 1) * LDSBUF;
        __bf16* nxt = lds + ((kt2 & 1) ^ 1) * LDSBUF;
        if (kt2 < ACT_DIM / BK - 1) {
            stage_async(Xh, Xl, Wh, Wl, nxt, tid, aRow0, bRow0, (kt2 + 1) * BK);
            WAIT_ASYNC(6);                  // current buffer's 6 async loads done
        } else {
            WAIT_ASYNC(0);
        }
        __syncthreads();

        v16bf a_hi[2], a_lo[2], b_hi[4], b_lo[4];
#pragma unroll
        for (int i = 0; i < 2; ++i) {
            int r = wm * 32 + i * 16 + lmn;
            const __bf16* p = cur + A_H + r * LDA + kh * 8;
            a_hi[i] = cat16(*(const v8bf*)p, *(const v8bf*)(p + 16));
            const __bf16* q = cur + A_L + r * LDA + kh * 8;
            a_lo[i] = cat16(*(const v8bf*)q, *(const v8bf*)(q + 16));
        }
#pragma unroll
        for (int j = 0; j < 4; ++j) {
            int r = wn * 64 + j * 16 + lmn;
            const __bf16* p = cur + B_H + r * LDA + kh * 8;
            b_hi[j] = cat16(*(const v8bf*)p, *(const v8bf*)(p + 16));
            const __bf16* q = cur + B_L + r * LDA + kh * 8;
            b_lo[j] = cat16(*(const v8bf*)q, *(const v8bf*)(q + 16));
        }
#pragma unroll
        for (int i = 0; i < 2; ++i)
#pragma unroll
            for (int j = 0; j < 4; ++j) {
                acc[i][j] = wmma_bf16(a_hi[i], b_hi[j], acc[i][j]);
                acc[i][j] = wmma_bf16(a_hi[i], b_lo[j], acc[i][j]);
                acc[i][j] = wmma_bf16(a_lo[i], b_hi[j], acc[i][j]);
            }
        __syncthreads();
    }

#pragma unroll
    for (int j = 0; j < 4; ++j) {
        int col = bRow0 + wn * 64 + j * 16 + lmn;
        float be = b_enc[col];
#pragma unroll
        for (int i = 0; i < 2; ++i)
#pragma unroll
            for (int v = 0; v < 8; ++v) {
                int rowg = aRow0 + wm * 32 + i * 16 + kh * 8 + v;
                float val = acc[i][j][v] + be;
                __builtin_nontemporal_store(val > 0.f ? val : 0.f,
                                            &pre[(long)rowg * DICT_SIZE + col]);
            }
    }
}

// ---------------- Kernel 1b: fallback GEMM (convert-in-kernel), used if ws too small ----------------
#define BM 128
#define BN 128

__global__ __launch_bounds__(256)
void sae_encode_gemm_conv(const float* __restrict__ x,
                          const float* __restrict__ W_enc,
                          const float* __restrict__ b_enc,
                          const float* __restrict__ b_dec,
                          float* __restrict__ pre) {
    __shared__ __bf16 Ah[BM * LDA];
    __shared__ __bf16 Al[BM * LDA];
    __shared__ __bf16 Bh[BN * LDA];
    __shared__ __bf16 Bl[BN * LDA];

    const int tid  = threadIdx.x;
    const int bn   = blockIdx.x;
    const int bm   = blockIdx.y;
    const int wave = tid >> 5;
    const int lane = tid & 31;
    const int wm   = wave & 3;
    const int wn   = wave >> 2;
    const int lmn  = lane & 15;
    const int kh   = lane >> 4;

    v8f acc[2][4];
#pragma unroll
    for (int i = 0; i < 2; ++i)
#pragma unroll
        for (int j = 0; j < 4; ++j) acc[i][j] = (v8f){0,0,0,0,0,0,0,0};

    for (int kt = 0; kt < ACT_DIM; kt += BK) {
#pragma unroll
        for (int i = 0; i < 4; ++i) {
            int linear = tid + i * 256;
            int r  = linear >> 3;
            int c4 = linear & 7;
            int k  = kt + c4 * 4;
            float4 xa = *(const float4*)&x[(long)(bm * BM + r) * ACT_DIM + k];
            float4 bd = *(const float4*)&b_dec[k];
            float4 wb = *(const float4*)&W_enc[(long)(bn * BN + r) * ACT_DIM + k];
            float av[4] = {xa.x - bd.x, xa.y - bd.y, xa.z - bd.z, xa.w - bd.w};
            float bv[4] = {wb.x, wb.y, wb.z, wb.w};
            uint2 ph, pl;
            int o = r * LDA + c4 * 4;
            split4(av, &ph, &pl);
            *(uint2*)&Ah[o] = ph;
            *(uint2*)&Al[o] = pl;
            split4(bv, &ph, &pl);
            *(uint2*)&Bh[o] = ph;
            *(uint2*)&Bl[o] = pl;
        }
        __syncthreads();

        v16bf a_hi[2], a_lo[2], b_hi[4], b_lo[4];
#pragma unroll
        for (int i = 0; i < 2; ++i) {
            int r = wm * 32 + i * 16 + lmn;
            const __bf16* p = &Ah[r * LDA + kh * 8];
            a_hi[i] = cat16(*(const v8bf*)p, *(const v8bf*)(p + 16));
            const __bf16* q = &Al[r * LDA + kh * 8];
            a_lo[i] = cat16(*(const v8bf*)q, *(const v8bf*)(q + 16));
        }
#pragma unroll
        for (int j = 0; j < 4; ++j) {
            int r = wn * 64 + j * 16 + lmn;
            const __bf16* p = &Bh[r * LDA + kh * 8];
            b_hi[j] = cat16(*(const v8bf*)p, *(const v8bf*)(p + 16));
            const __bf16* q = &Bl[r * LDA + kh * 8];
            b_lo[j] = cat16(*(const v8bf*)q, *(const v8bf*)(q + 16));
        }
#pragma unroll
        for (int i = 0; i < 2; ++i)
#pragma unroll
            for (int j = 0; j < 4; ++j) {
                acc[i][j] = wmma_bf16(a_hi[i], b_hi[j], acc[i][j]);
                acc[i][j] = wmma_bf16(a_hi[i], b_lo[j], acc[i][j]);
                acc[i][j] = wmma_bf16(a_lo[i], b_hi[j], acc[i][j]);
            }
        __syncthreads();
    }

#pragma unroll
    for (int j = 0; j < 4; ++j) {
        int col = bn * BN + wn * 64 + j * 16 + lmn;
        float be = b_enc[col];
#pragma unroll
        for (int i = 0; i < 2; ++i)
#pragma unroll
            for (int v = 0; v < 8; ++v) {
                int rowg = bm * BM + wm * 32 + i * 16 + kh * 8 + v;
                float val = acc[i][j][v] + be;
                __builtin_nontemporal_store(val > 0.f ? val : 0.f,
                                            &pre[(long)rowg * DICT_SIZE + col]);
            }
    }
}

// ---------------- Kernel 2: per-row top-32 + sparsify + sparse decode ----------------
#define TPB 256
#define EPT4 (DICT_SIZE / TPB / 4)   // 24 float4 per thread

__global__ __launch_bounds__(TPB)
void sae_topk_decode(const float* __restrict__ W_dec,
                     const float* __restrict__ b_dec,
                     float* __restrict__ recon,
                     float* __restrict__ enc) {
    __shared__ float srow[DICT_SIZE];    // 96 KB
    __shared__ float s_wval[8];
    __shared__ int   s_widx[8];
    __shared__ float s_tval[TOPK];
    __shared__ int   s_tidx[TOPK];
    __shared__ int   s_bidx;

    const int row  = blockIdx.x;
    const int tid  = threadIdx.x;
    const int lane = tid & 31;
    const int wv   = tid >> 5;
    float* erow = enc + (long)row * DICT_SIZE;

    // load row (post-relu, >= 0) into LDS, track per-thread local max
    float mv = -1.0f; int mi = -1;
    for (int i = 0; i < EPT4; ++i) {
        int d4 = tid + i * TPB;
        f4 v = __builtin_nontemporal_load((const f4*)erow + d4);
        ((f4*)srow)[d4] = v;
        int base = d4 << 2;
        if (v.x > mv) { mv = v.x; mi = base; }
        if (v.y > mv) { mv = v.y; mi = base + 1; }
        if (v.z > mv) { mv = v.z; mi = base + 2; }
        if (v.w > mv) { mv = v.w; mi = base + 3; }
    }
    __syncthreads();

    // 32 iterations of block-wide argmax (ties -> lower index, matches lax.top_k)
    for (int j = 0; j < TOPK; ++j) {
        float v = mv; int d = mi;
#pragma unroll
        for (int off = 16; off > 0; off >>= 1) {
            float ov = __shfl_down(v, off);
            int   od = __shfl_down(d, off);
            if (ov > v || (ov == v && od != -1 && (d == -1 || od < d))) { v = ov; d = od; }
        }
        if (lane == 0) { s_wval[wv] = v; s_widx[wv] = d; }
        __syncthreads();
        if (tid == 0) {
            float bv = s_wval[0]; int bd = s_widx[0];
            for (int w = 1; w < 8; ++w) {
                float ov = s_wval[w]; int od = s_widx[w];
                if (ov > bv || (ov == bv && od != -1 && (bd == -1 || od < bd))) { bv = ov; bd = od; }
            }
            s_tval[j] = bv; s_tidx[j] = bd; s_bidx = bd;
        }
        __syncthreads();
        int bd = s_bidx;
        if (((bd >> 2) & (TPB - 1)) == tid) {
            srow[bd] = -(float)(j + 1);     // rank marker (live values are >= 0)
            mv = -1.0f; mi = -1;            // rescan this thread's slice
            for (int i = 0; i < EPT4; ++i) {
                int base = (tid + i * TPB) << 2;
#pragma unroll
                for (int c = 0; c < 4; ++c) {
                    float vv = srow[base + c];
                    if (vv > mv) { mv = vv; mi = base + c; }
                }
            }
        }
        __syncthreads();
    }

    // write sparse encoded row: rank markers -> top value, everything else -> 0
    for (int i = 0; i < EPT4; ++i) {
        int d4 = tid + i * TPB;
        f4 v = ((const f4*)srow)[d4];
        f4 o;
        o.x = (v.x < 0.f) ? s_tval[(int)(-v.x) - 1] : 0.f;
        o.y = (v.y < 0.f) ? s_tval[(int)(-v.y) - 1] : 0.f;
        o.z = (v.z < 0.f) ? s_tval[(int)(-v.z) - 1] : 0.f;
        o.w = (v.w < 0.f) ? s_tval[(int)(-v.w) - 1] : 0.f;
        __builtin_nontemporal_store(o, (f4*)erow + d4);
    }

    // sparse decode: recon[a] = b_dec[a] + sum_j tval[j] * W_dec[a, tidx[j]]
    for (int a = tid; a < ACT_DIM; a += TPB) {
        float s = b_dec[a];
        const float* wrow = W_dec + (long)a * DICT_SIZE;
#pragma unroll
        for (int j = 0; j < TOPK; ++j) s += s_tval[j] * wrow[s_tidx[j]];
        recon[(long)row * ACT_DIM + a] = s;
    }
}

// ---------------- launch ----------------
extern "C" void kernel_launch(void* const* d_in, const int* in_sizes, int n_in,
                              void* d_out, int out_size, void* d_ws, size_t ws_size,
                              hipStream_t stream) {
    const float* x     = (const float*)d_in[0];
    const float* W_enc = (const float*)d_in[1];
    const float* b_enc = (const float*)d_in[2];
    const float* W_dec = (const float*)d_in[3];
    const float* b_dec = (const float*)d_in[4];

    float* recon = (float*)d_out;                           // [8192, 768]
    float* enc   = (float*)d_out + (long)NROWS * ACT_DIM;   // [8192, 24576]

    const long XE = (long)NROWS * ACT_DIM;       // 6,291,456
    const long WE = (long)DICT_SIZE * ACT_DIM;   // 18,874,368
    const size_t ws_need = (size_t)(2 * XE + 2 * WE) * sizeof(__bf16);  // 100,663,296 B

    if (ws_size >= ws_need) {
        __bf16* Xh = (__bf16*)d_ws;
        __bf16* Xl = Xh + XE;
        __bf16* Wh = Xl + XE;
        __bf16* Wl = Wh + WE;
        cvt_split_x<<<(unsigned)(XE / 4 / 256), 256, 0, stream>>>(x, b_dec, Xh, Xl);
        cvt_split_w<<<(unsigned)(WE / 4 / 256), 256, 0, stream>>>(W_enc, Wh, Wl);
        dim3 g1(DICT_SIZE / BN2, NROWS / BM2);   // 192 x 32
        sae_encode_gemm_async<<<g1, 512, 0, stream>>>(Xh, Xl, Wh, Wl, b_enc, enc);
    } else {
        dim3 g1(DICT_SIZE / BN, NROWS / BM);     // 192 x 64
        sae_encode_gemm_conv<<<g1, 256, 0, stream>>>(x, W_enc, b_enc, b_dec, enc);
    }
    sae_topk_decode<<<NROWS, TPB, 0, stream>>>(W_dec, b_dec, recon, enc);
}
